// GIN_29386166239460
// MI455X (gfx1250) — matrix operand order repaired and checked by reading, not compiled, not run on hardware
//
#include <hip/hip_runtime.h>
#include <hip/hip_bf16.h>

typedef __attribute__((ext_vector_type(16))) _Float16 v16h;
typedef __attribute__((ext_vector_type(8)))  _Float16 v8h;
typedef __attribute__((ext_vector_type(4)))  _Float16 v4h;
typedef __attribute__((ext_vector_type(8)))  float    v8f;

#define DIM 128

// ---------------------------------------------------------------------------
// agg = x  (float4 granularity). Seeding agg with x makes the scatter produce
// (1+eps)*x + sum  directly (EPS = 0), killing one 51.2MB read pass per layer.
__global__ __launch_bounds__(256) void copy_f4(
    const float4* __restrict__ x, float4* __restrict__ agg, int n4)
{
    int t = blockIdx.x * blockDim.x + threadIdx.x;
    if (t < n4) agg[t] = x[t];
}

// ---------------------------------------------------------------------------
// Edge-parallel scatter-add: one wave per edge, lanes stripe the 128 features
// as float4 (coalesced 512B per wave). src/dst loads are wave-uniform.
__global__ __launch_bounds__(256) void scatter_add(
    const float* __restrict__ x, const int* __restrict__ src,
    const int* __restrict__ dst, float* __restrict__ agg, int nEdges)
{
    int t = blockIdx.x * blockDim.x + threadIdx.x;
    int e = t >> 5;
    if (e >= nEdges) return;
    int f = (t & 31) << 2;                       // lane*4 features
    int s = src[e];
    int d = dst[e];
    float4 v = *(const float4*)(x + (size_t)s * DIM + f);
    float* ap = agg + (size_t)d * DIM + f;
    atomicAdd(ap + 0, v.x);
    atomicAdd(ap + 1, v.y);
    atomicAdd(ap + 2, v.z);
    atomicAdd(ap + 3, v.w);
}

// ---------------------------------------------------------------------------
// a_f16 = f16(agg)   (agg already holds x + sum)
__global__ __launch_bounds__(256) void convert_f16(
    const float* __restrict__ agg, _Float16* __restrict__ out, int n4)
{
    int t = blockIdx.x * blockDim.x + threadIdx.x;
    if (t >= n4) return;
    float4 g = ((const float4*)agg)[t];
    v4h h;
    h[0] = (_Float16)g.x;
    h[1] = (_Float16)g.y;
    h[2] = (_Float16)g.z;
    h[3] = (_Float16)g.w;
    ((v4h*)out)[t] = h;
}

// ---------------------------------------------------------------------------
// Pack W [128 x 128] f32 row-major (K x N) into a per-lane B-operand layout
// for V_WMMA_F32_16X16X32_F16 in contiguous k order (same as A in the GEMM):
//   for tile (nt in 0..7, kk in 0..3), lane L holds column n = nt*16+(L&15),
//   halfs j=0..15 = W[kk*32 + ((L>=16)?16:0) + j][n], stored contiguously.
__global__ __launch_bounds__(256) void pack_w(
    const float* __restrict__ W, _Float16* __restrict__ Bp)
{
    int t = blockIdx.x * blockDim.x + threadIdx.x;   // 8*4*32 = 1024 threads
    if (t >= 8 * 4 * 32) return;
    int lane = t & 31;
    int kk   = (t >> 5) & 3;
    int nt   = t >> 7;
    int n     = (nt << 4) + (lane & 15);
    int kbase = (kk << 5) + ((lane >> 4) << 4);
    _Float16* dstp = Bp + (size_t)t * 16;
#pragma unroll
    for (int j = 0; j < 16; ++j)
        dstp[j] = (_Float16)W[(size_t)(kbase + j) * DIM + n];
}

// ---------------------------------------------------------------------------
// GEMM: C[M x 128] = (relu?)( A[M x 128] @ W + b )
// One wave computes one 16x16 output tile; K=128 via 4x WMMA f32_16x16x32_f16.
template <bool RELU>
__global__ __launch_bounds__(256) void gin_gemm_wmma(
    const _Float16* __restrict__ A,   // [M x 128] row-major f16
    const _Float16* __restrict__ Bp,  // packed [nt][kk][lane][16]
    const float*    __restrict__ bias,
    float*          __restrict__ C,   // [M x 128] f32
    int M)
{
    int wave = (blockIdx.x * blockDim.x + threadIdx.x) >> 5;
    int lane = threadIdx.x & 31;
    int mt = wave >> 3;                 // 8 N-tiles per M-tile
    int nt = wave & 7;
    if (mt >= (M >> 4)) return;         // exact fit for M=100000 (uniform)

    int lm    = lane & 15;
    int hi    = lane >> 4;              // 0 or 1
    int m     = (mt << 4) + lm;         // A row for this lane
    int n     = (nt << 4) + lm;         // output column for this lane
    int kbA   = hi << 3;                // A K sub-offset: 0 or 8

    const _Float16* arow  = A + (size_t)m * DIM;
    const _Float16* bbase = Bp + ((size_t)(nt * 4) * 32 + lane) * 16;

    float bn = bias[n];

    v8f c = {};
#pragma unroll
    for (int kk = 0; kk < 4; ++kk) {
        int k0 = kk << 5;
        // A 16x32 layout (matches packed B): halfs 0..15 = K=2*kbA+0..15
        v8h a_lo = *(const v8h*)(arow + k0 + 2 * kbA);
        v8h a_hi = *(const v8h*)(arow + k0 + 2 * kbA + 8);
        v16h a;
#pragma unroll
        for (int j = 0; j < 8; ++j) { a[j] = a_lo[j]; a[j + 8] = a_hi[j]; }
        v16h b = *(const v16h*)(bbase + (size_t)kk * (32 * 16));
        c = __builtin_amdgcn_wmma_f32_16x16x32_f16(
                /*neg_a=*/false, a, /*neg_b=*/false, b,
                /*c_mod=*/(short)0, c, /*reuse_a=*/false, /*reuse_b=*/false);
    }

#pragma unroll
    for (int v = 0; v < 8; ++v) {       // C/D layout: VGPR v -> row v + 8*hi
        float o = c[v] + bn;
        if (RELU) o = fmaxf(o, 0.0f);   // single v_max_f32
        int mrow = (mt << 4) + v + (hi << 3);
        C[(size_t)mrow * DIM + n] = o;
    }
}

// ---------------------------------------------------------------------------
// Row-wise log_softmax, one wave (32 lanes x float4) per 128-wide row, in place.
__global__ __launch_bounds__(256) void log_softmax_k(
    const float* __restrict__ h, float* __restrict__ out, int nNodes)
{
    int wave = (blockIdx.x * blockDim.x + threadIdx.x) >> 5;
    int lane = threadIdx.x & 31;
    if (wave >= nNodes) return;
    float4 v = ((const float4*)(h + (size_t)wave * DIM))[lane];

    float mx = fmaxf(fmaxf(v.x, v.y), fmaxf(v.z, v.w));
#pragma unroll
    for (int off = 16; off >= 1; off >>= 1)
        mx = fmaxf(mx, __shfl_xor(mx, off, 32));

    float s = __expf(v.x - mx) + __expf(v.y - mx)
            + __expf(v.z - mx) + __expf(v.w - mx);
#pragma unroll
    for (int off = 16; off >= 1; off >>= 1)
        s += __shfl_xor(s, off, 32);

    float lse = mx + __logf(s);
    float4 o = make_float4(v.x - lse, v.y - lse, v.z - lse, v.w - lse);
    ((float4*)(out + (size_t)wave * DIM))[lane] = o;
}

// ---------------------------------------------------------------------------
extern "C" void kernel_launch(void* const* d_in, const int* in_sizes, int n_in,
                              void* d_out, int out_size, void* d_ws, size_t ws_size,
                              hipStream_t stream)
{
    const float* x  = (const float*)d_in[0];
    const int*   ei = (const int*)  d_in[1];
    const float* W1 = (const float*)d_in[2];
    const float* b1 = (const float*)d_in[3];
    const float* W2 = (const float*)d_in[4];
    const float* b2 = (const float*)d_in[5];
    float* out = (float*)d_out;

    const int N = in_sizes[0] / DIM;      // 100000
    const int E = in_sizes[1] / 2;        // 1600000
    const int* src = ei;
    const int* dst = ei + E;

    char* ws = (char*)d_ws;
    const size_t featBytes = (size_t)N * DIM * sizeof(float);   // 51.2 MB
    float*    agg   = (float*)ws;
    float*    h1    = (float*)(ws + featBytes);
    _Float16* af16  = (_Float16*)(ws + 2 * featBytes);
    _Float16* wpack = (_Float16*)(ws + 2 * featBytes + (size_t)N * DIM * sizeof(_Float16));

    const int n4      = N * DIM / 4;                 // 3.2M float4
    const int zBlocks = (n4 + 255) / 256;
    const int sBlocks = (E * 32 + 255) / 256;        // wave per edge
    const int gBlocks = (N / 16);                    // 50000 tiles / 8 waves per block
    const int lBlocks = (N * 32 + 255) / 256;

    // ---- Layer 1: agg = x + scatter(x); h1 = relu(agg @ W1 + b1)
    copy_f4      <<<zBlocks, 256, 0, stream>>>((const float4*)x, (float4*)agg, n4);
    scatter_add  <<<sBlocks, 256, 0, stream>>>(x, src, dst, agg, E);
    convert_f16  <<<zBlocks, 256, 0, stream>>>(agg, af16, n4);
    pack_w       <<<4,       256, 0, stream>>>(W1, wpack);
    gin_gemm_wmma<true> <<<gBlocks, 256, 0, stream>>>(af16, wpack, b1, h1, N);

    // ---- Layer 2: agg = h1 + scatter(h1); out = agg @ W2 + b2
    copy_f4      <<<zBlocks, 256, 0, stream>>>((const float4*)h1, (float4*)agg, n4);
    scatter_add  <<<sBlocks, 256, 0, stream>>>(h1, src, dst, agg, E);
    convert_f16  <<<zBlocks, 256, 0, stream>>>(agg, af16, n4);
    pack_w       <<<4,       256, 0, stream>>>(W2, wpack);
    gin_gemm_wmma<false><<<gBlocks, 256, 0, stream>>>(af16, wpack, b2, out, N);

    // ---- log_softmax rows, in place on d_out
    log_softmax_k<<<lBlocks, 256, 0, stream>>>(out, out, N);
}
